// BlockFast_84670985273588
// MI455X (gfx1250) — compile-verified, matrix operand order checked by reading, not run on hardware
//
#include <hip/hip_runtime.h>
#include <hip/hip_bf16.h>
#include <math.h>
#include <stdint.h>

// ---------------- problem constants (from reference) ----------------
#define N_TOK   65536      // B*T = 8*8192
#define D_IN    1024
#define H_DIM   1024
#define D_OUT   1024
#define D_ADDR  64
#define L_EXP   12
#define K_TOP   3
#define TAU_F   1.0f
#define EPS_F   1e-8f

typedef __attribute__((ext_vector_type(16))) __bf16 bf16x16;
typedef __attribute__((ext_vector_type(8)))  __bf16 bf16x8;
typedef __attribute__((ext_vector_type(4)))  __bf16 bf16x4;
typedef __attribute__((ext_vector_type(8)))  float  f32x8;

// LDS layout for the staged A tile: 64 rows x 1024 bf16, 16B row padding
// (row stride 2064B -> bank rotation of 4 per row -> conflict-free half-wave
//  ds_load_b128 fragment reads).
#define LDS_ROW   (2048 + 16)
#define LDS_BYTES (64 * LDS_ROW)      // 132096 B, dynamic LDS (<< 320KB/WGP)

// ---------------- fp32 -> bf16 conversion (vectorized, grid-stride) ----------------
__global__ void cvt_f32_to_bf16_x4(const float* __restrict__ src,
                                   __bf16* __restrict__ dst, int n4) {
  int i = blockIdx.x * blockDim.x + threadIdx.x;
  int stride = gridDim.x * blockDim.x;
  for (; i < n4; i += stride) {
    float4 v = ((const float4*)src)[i];
    bf16x4 o;
    o[0] = (__bf16)v.x; o[1] = (__bf16)v.y;
    o[2] = (__bf16)v.z; o[3] = (__bf16)v.w;
    ((bf16x4*)dst)[i] = o;
  }
}

// ---------------- routing: one 64-thread block per token ----------------
__global__ void routing_kernel(const float* __restrict__ x,
                               const float* __restrict__ Pw,   // [64, 1024]
                               const float* __restrict__ U1,   // [12, 64]
                               const float* __restrict__ U2,
                               const float* __restrict__ U3,
                               float* __restrict__ C1,         // [N, 12]
                               float* __restrict__ C2,
                               float* __restrict__ C3) {
  __shared__ float xs[D_IN];
  __shared__ float xa[D_ADDR];
  __shared__ float zz[3][L_EXP];
  const int n = blockIdx.x;
  const int t = threadIdx.x;                 // 0..63
  const float* xr = x + (size_t)n * D_IN;

  for (int k = t * 4; k < D_IN; k += 64 * 4)
    *(float4*)(xs + k) = *(const float4*)(xr + k);
  __syncthreads();

  {
    const float* pr = Pw + (size_t)t * D_IN;
    float acc = 0.f;
    for (int k = 0; k < D_IN; k += 4) {
      float4 p = *(const float4*)(pr + k);
      acc += xs[k] * p.x + xs[k + 1] * p.y + xs[k + 2] * p.z + xs[k + 3] * p.w;
    }
    xa[t] = acc;
  }
  __syncthreads();

  if (t < 3 * L_EXP) {
    int r = t / L_EXP, e = t % L_EXP;
    const float* U = (r == 0 ? U1 : (r == 1 ? U2 : U3)) + e * D_ADDR;
    float acc = 0.f;
    for (int k = 0; k < D_ADDR; ++k) acc += xa[k] * U[k];
    zz[r][e] = acc;
  }
  __syncthreads();

  if (t < 3) {
    float* Cr = (t == 0 ? C1 : (t == 1 ? C2 : C3)) + (size_t)n * L_EXP;
    float z[L_EXP];
    for (int e = 0; e < L_EXP; ++e) z[e] = zz[t][e];
    int   idx[K_TOP];
    float val[K_TOP];
    for (int j = 0; j < K_TOP; ++j) {
      int bi = -1; float bv = -1e30f;
      for (int e = 0; e < L_EXP; ++e) {
        bool taken = false;
        for (int q = 0; q < j; ++q) taken |= (idx[q] == e);
        if (!taken && z[e] > bv) { bv = z[e]; bi = e; }
      }
      idx[j] = bi; val[j] = bv;
    }
    const float inv_tau = 1.0f / (TAU_F + EPS_F);
    float m = val[0], s = 0.f, w[K_TOP];
    for (int j = 0; j < K_TOP; ++j) { w[j] = __expf((val[j] - m) * inv_tau); s += w[j]; }
    float inv_s = 1.0f / s;
    for (int e = 0; e < L_EXP; ++e) Cr[e] = 0.f;
    for (int j = 0; j < K_TOP; ++j) Cr[idx[j]] = w[j] * inv_s;
  }
}

// ---------------- WMMA mixture GEMM with async-LDS-staged A ----------------
// Block = 4 waves = 64 rows x 128 cols of output; each wave owns 64x32.
// A (X tile, 64x1024 bf16 = 128KB) is DMA'd once per block into dynamic LDS
// via GLOBAL_LOAD_ASYNC_TO_LDS_B128 (ASYNCcnt-tracked, no VGPR data) and
// re-read per expert via ds_load_b128 (DS pipe co-executes with the global
// B weight stream). Per expert: K-loop of v_wmma_f32_16x16x32_bf16, then
// scale rows by C[row,e] into fp32 final accumulators (exact by linearity).
// Next expert's B rows are prefetched (global_prefetch_b8).
//
// Fragment layout (16-bit A/B 16x32, ISA 7.12.2):
//   lane<16 : M/N = lane,    elems 0..7 = K 0..7,  8..15 = K 16..23
//   lane>=16: M/N = lane-16, elems 0..7 = K 8..15, 8..15 = K 24..31
// C/D: VGPR v, lane l -> row = v + 8*(l>=16), col = l%16.
template <int STAGE>   // 0: GELU -> bf16 H buffer ; 1: fp32 out + routed bias
__global__ __launch_bounds__(128, 1)
void moe_mixture_wmma(const __bf16* __restrict__ X,   // [N, 1024] bf16
                      const __bf16* __restrict__ W,   // [12, OUTD, 1024] bf16
                      const float*  __restrict__ C,   // [N, 12]
                      const float*  __restrict__ C3,  // [N, 12] (stage 1)
                      const float*  __restrict__ b2,  // [12, OUTD] (stage 1)
                      __bf16* __restrict__ outb,      // stage 0
                      float*  __restrict__ outf) {    // stage 1
  extern __shared__ char smem[];                      // LDS_BYTES
  const int KDIM = 1024, OUTD = 1024;
  const int tid  = threadIdx.x;
  const int lane = tid & 31;
  const int wid  = tid >> 5;
  const int rowband = blockIdx.x / (OUTD / 128);
  const int colgrp  = blockIdx.x % (OUTD / 128);
  const int rowbase = rowband * 64;
  const int colbase = colgrp * 128 + wid * 32;
  const int lr = lane & 15;
  const int hi = lane >> 4;
  const int kadj = hi * 8;                            // per-lane K sub-offset

  // ---- async DMA the A tile into LDS: 64 rows x 2048B, one row per iter ----
  // VDST VGPR = per-lane LDS byte offset (LDS aperture truncates generic
  // addresses to addr[31:0], so low 32 bits of the generic pointer are the
  // DS offset the hardware adds to LDS_BASE). VADDR = 64-bit global address.
  for (int r = 0; r < 64; ++r) {
    const __bf16* g = X + (size_t)(rowbase + r) * KDIM + tid * 8;
    unsigned lds_off = (unsigned)(uintptr_t)(smem + r * LDS_ROW + tid * 16);
    asm volatile("global_load_async_to_lds_b128 %0, %1, off"
                 :: "v"(lds_off), "v"((uint64_t)(uintptr_t)g)
                 : "memory");
  }
#if __has_builtin(__builtin_amdgcn_s_wait_asynccnt)
  __builtin_amdgcn_s_wait_asynccnt(0);
#else
  asm volatile("s_wait_asynccnt 0x0" ::: "memory");
#endif
  __syncthreads();

  f32x8 facc[4][2] = {};

  for (int e = 0; e < L_EXP; ++e) {
    f32x8 eacc[4][2] = {};
    const __bf16* We = W + (size_t)e * OUTD * KDIM;
    const __bf16* Wn = W + (size_t)(e + 1 < L_EXP ? e + 1 : e) * OUTD * KDIM;

    #pragma unroll 2
    for (int k0 = 0; k0 < KDIM; k0 += 32) {
      // prefetch next expert's weight rows at this K position (1 line / row)
      __builtin_prefetch(Wn + (size_t)(colbase + lane) * KDIM + k0, 0, 1);

      bf16x16 a[4], b[2];
      #pragma unroll
      for (int mt = 0; mt < 4; ++mt) {
        const char* p = smem + (mt * 16 + lr) * LDS_ROW + (k0 + kadj) * 2;
        union { bf16x16 v; bf16x8 h[2]; } u;
        u.h[0] = *(const bf16x8*)p;          // K 0..7   (or 8..15)
        u.h[1] = *(const bf16x8*)(p + 32);   // K 16..23 (or 24..31)
        a[mt] = u.v;
      }
      #pragma unroll
      for (int nt = 0; nt < 2; ++nt) {
        const __bf16* p = We + (size_t)(colbase + nt * 16 + lr) * KDIM + k0 + kadj;
        union { bf16x16 v; bf16x8 h[2]; } u;
        u.h[0] = *(const bf16x8*)p;
        u.h[1] = *(const bf16x8*)(p + 16);
        b[nt] = u.v;
      }
      #pragma unroll
      for (int mt = 0; mt < 4; ++mt)
        #pragma unroll
        for (int nt = 0; nt < 2; ++nt)
          eacc[mt][nt] = __builtin_amdgcn_wmma_f32_16x16x32_bf16(
              false, a[mt], false, b[nt], (short)0, eacc[mt][nt], false, false);
    }

    // fold expert tile into final accumulators, scaled by C[row, e]
    #pragma unroll
    for (int mt = 0; mt < 4; ++mt) {
      float s[8];
      #pragma unroll
      for (int v = 0; v < 8; ++v)
        s[v] = C[(size_t)(rowbase + mt * 16 + hi * 8 + v) * L_EXP + e];
      #pragma unroll
      for (int nt = 0; nt < 2; ++nt)
        #pragma unroll
        for (int v = 0; v < 8; ++v)
          facc[mt][nt][v] += s[v] * eacc[mt][nt][v];
    }
  }

  if (STAGE == 0) {
    // exact GELU, store bf16
    #pragma unroll
    for (int mt = 0; mt < 4; ++mt)
      #pragma unroll
      for (int nt = 0; nt < 2; ++nt)
        #pragma unroll
        for (int v = 0; v < 8; ++v) {
          float g = facc[mt][nt][v];
          float y = 0.5f * g * (1.0f + erff(g * 0.70710678118654752f));
          size_t row = (size_t)(rowbase + mt * 16 + hi * 8 + v);
          outb[row * OUTD + colbase + nt * 16 + lr] = (__bf16)y;
        }
  } else {
    // routed bias: out[n,h] += sum_e C3[n,e] * b2[e,h]
    float bb[L_EXP][2];
    #pragma unroll
    for (int e = 0; e < L_EXP; ++e)
      #pragma unroll
      for (int nt = 0; nt < 2; ++nt)
        bb[e][nt] = b2[(size_t)e * OUTD + colbase + nt * 16 + lr];
    #pragma unroll
    for (int mt = 0; mt < 4; ++mt) {
      #pragma unroll
      for (int v = 0; v < 8; ++v) {
        const float* c3 = C3 + (size_t)(rowbase + mt * 16 + hi * 8 + v) * L_EXP;
        float a0 = 0.f, a1 = 0.f;
        #pragma unroll
        for (int e = 0; e < L_EXP; ++e) {
          float s = c3[e];
          a0 += s * bb[e][0];
          a1 += s * bb[e][1];
        }
        facc[mt][0][v] += a0;
        facc[mt][1][v] += a1;
      }
    }
    #pragma unroll
    for (int mt = 0; mt < 4; ++mt)
      #pragma unroll
      for (int nt = 0; nt < 2; ++nt)
        #pragma unroll
        for (int v = 0; v < 8; ++v) {
          size_t row = (size_t)(rowbase + mt * 16 + hi * 8 + v);
          outf[row * OUTD + colbase + nt * 16 + lr] = facc[mt][nt][v];
        }
  }
}

// ---------------- launcher ----------------
extern "C" void kernel_launch(void* const* d_in, const int* in_sizes, int n_in,
                              void* d_out, int out_size, void* d_ws, size_t ws_size,
                              hipStream_t stream) {
  const float* x  = (const float*)d_in[0];
  const float* Pw = (const float*)d_in[1];
  const float* U1 = (const float*)d_in[2];
  const float* U2 = (const float*)d_in[3];
  const float* U3 = (const float*)d_in[4];
  const float* W1 = (const float*)d_in[5];
  const float* W2 = (const float*)d_in[6];
  const float* b2 = (const float*)d_in[7];

  char* ws = (char*)d_ws;
  size_t off = 0;
  __bf16* Xb  = (__bf16*)(ws + off); off += (size_t)N_TOK * D_IN  * 2;
  __bf16* Hb  = (__bf16*)(ws + off); off += (size_t)N_TOK * H_DIM * 2;
  __bf16* W1b = (__bf16*)(ws + off); off += (size_t)L_EXP * H_DIM * D_IN  * 2;
  __bf16* W2b = (__bf16*)(ws + off); off += (size_t)L_EXP * D_OUT * H_DIM * 2;
  float*  C1  = (float*)(ws + off);  off += (size_t)N_TOK * L_EXP * 4;
  float*  C2  = (float*)(ws + off);  off += (size_t)N_TOK * L_EXP * 4;
  float*  C3  = (float*)(ws + off);  off += (size_t)N_TOK * L_EXP * 4;

  // bf16 conversions
  cvt_f32_to_bf16_x4<<<8192, 256, 0, stream>>>(x,  Xb,  (N_TOK * D_IN) / 4);
  cvt_f32_to_bf16_x4<<<4096, 256, 0, stream>>>(W1, W1b, (L_EXP * H_DIM * D_IN) / 4);
  cvt_f32_to_bf16_x4<<<4096, 256, 0, stream>>>(W2, W2b, (L_EXP * D_OUT * H_DIM) / 4);

  // routing (projection + 3 routers + top-3 softmax)
  routing_kernel<<<N_TOK, 64, 0, stream>>>(x, Pw, U1, U2, U3, C1, C2, C3);

  // blocks: (N/64 row bands) x (OUTD/128 col groups), 4 waves/block
  const int gemm_blocks = (N_TOK / 64) * (H_DIM / 128);   // 8192

  // layer 1: h = gelu(mixture(x, C1, W1)) -> bf16
  moe_mixture_wmma<0><<<gemm_blocks, 128, LDS_BYTES, stream>>>(
      Xb, W1b, C1, nullptr, nullptr, Hb, nullptr);

  // layer 2: y = mixture(h, C2, W2) + C3 @ b2 -> fp32 out
  moe_mixture_wmma<1><<<gemm_blocks, 128, LDS_BYTES, stream>>>(
      Hb, W2b, C2, C3, b2, nullptr, (float*)d_out);
}
